// CostumeQuantizer_35467839930386
// MI455X (gfx1250) — matrix operand order repaired and checked by reading, not compiled, not run on hardware
//
#include <hip/hip_runtime.h>
#include <hip/hip_bf16.h>

// ---------------------------------------------------------------------------
// Residual VQ codec layer for MI455X (gfx1250, wave32, WMMA + async-to-LDS).
// All GEMM stages: bf16 operands, v_wmma_f32_16x16x32_bf16, f32 accumulate.
// Operands pre-converted to bf16 once so the GEMM mainloop is a pure
// async-copy -> WMMA pipeline (no f32->bf16 VALU work per tile).
// ---------------------------------------------------------------------------

typedef __bf16 bf16_t;
typedef __attribute__((ext_vector_type(16))) __bf16 bf16x16;
typedef __attribute__((ext_vector_type(8)))  __bf16 bf16x8;
typedef __attribute__((ext_vector_type(8)))  float  f32x8;
typedef __attribute__((ext_vector_type(4)))  int    int4v;

static constexpr int Bb   = 16;
static constexpr int Tt   = 2000;
static constexpr int DIN  = 1024;
static constexpr int DC   = 512;
static constexpr int NQ   = 8;
static constexpr int KC   = 1024;          // codebook entries
static constexpr int MR   = Bb * Tt;       // 32000 rows
static constexpr float CODEC_RANGE = 8.0f;

#define BM 128
#define BN 128
#define BK 32

// --------------------------- async copy to LDS ------------------------------
#if defined(__has_builtin)
#  if __has_builtin(__builtin_amdgcn_global_load_async_to_lds_b128) && \
      __has_builtin(__builtin_amdgcn_s_wait_asynccnt)
#    define USE_ASYNC_LDS 1
#  endif
#endif
#ifndef USE_ASYNC_LDS
#  define USE_ASYNC_LDS 0
#endif

#define AS_GLOBAL __attribute__((address_space(1)))
#define AS_LDS    __attribute__((address_space(3)))

__device__ __forceinline__ void copy16_to_lds(bf16_t* lds_dst, const bf16_t* gsrc) {
#if USE_ASYNC_LDS
  // signature (from probe diagnostics): (int4 AS1*, int4 AS3*, imm offset, imm cpol)
  __builtin_amdgcn_global_load_async_to_lds_b128(
      (AS_GLOBAL int4v*)(unsigned long long)(uintptr_t)gsrc,
      (AS_LDS int4v*)lds_dst, 0, 0);
#else
  *(bf16x8*)lds_dst = *(const bf16x8*)gsrc;
#endif
}

// ---------------------------------------------------------------------------
// WMMA GEMM:  A [M][K] bf16 row-major,  B [N][K] bf16 row-major (= B^T).
// Block tile 128x128, BK=32, 8 waves; wave = 32M x 64N = 2x4 WMMA tiles.
// Double-buffered LDS staging via GLOBAL_LOAD_ASYNC_TO_LDS_B128 (ASYNCcnt).
// MODE 0: C = tanh(acc+aux[n])*8 ; C2 = 0 ; Cb = (bf16)C   (input proj)
// MODE 1: C = aux[n] - 2*acc                               (VQ distances)
// MODE 2: C = acc + aux[n]                                 (output proj)
// ---------------------------------------------------------------------------
template <int MODE>
__global__ __launch_bounds__(256)
void gemm_wmma_kernel(const bf16_t* __restrict__ A, const bf16_t* __restrict__ Bm,
                      const float* __restrict__ aux,
                      float* __restrict__ C, float* __restrict__ C2,
                      bf16_t* __restrict__ Cb, int M, int N, int Kdim) {
  __shared__ __align__(32) bf16_t As[2][BM * BK];   // [m][k]
  __shared__ __align__(32) bf16_t Bs[2][BN * BK];   // [n][k]

  const int tid  = threadIdx.x;
  const int lane = tid & 31;
  const int wave = tid >> 5;
  const int wm   = wave & 3;     // 4 wave-rows  -> 32 M each
  const int wn   = wave >> 2;    // 2 wave-cols  -> 64 N each
  const int m0   = blockIdx.x * BM;
  const int n0   = blockIdx.y * BN;
  const int half = lane >> 4;

  f32x8 acc[2][4] = {};

  // stage one 128x32 bf16 tile pair: 512 x b128 chunks each, 2 per thread
  auto stage = [&](int buf, int k0) {
#pragma unroll
    for (int r = 0; r < 2; ++r) {
      const int c   = tid + r * 256;   // chunk id 0..511
      const int row = c >> 2;          // 4 chunks (of 8 bf16) per 32-wide row
      const int c4  = c & 3;
      copy16_to_lds(&As[buf][row * BK + c4 * 8],
                    A + (size_t)(m0 + row) * Kdim + k0 + c4 * 8);
      copy16_to_lds(&Bs[buf][row * BK + c4 * 8],
                    Bm + (size_t)(n0 + row) * Kdim + k0 + c4 * 8);
    }
  };

  const int nsteps = Kdim / BK;
  stage(0, 0);

  for (int i = 0; i < nsteps; ++i) {
    if (i + 1 < nsteps) {
      stage((i + 1) & 1, (i + 1) * BK);      // prefetch next tile pair
#if USE_ASYNC_LDS
      __builtin_amdgcn_s_wait_asynccnt(4);   // 4 new copies stay in flight
#endif
    } else {
#if USE_ASYNC_LDS
      __builtin_amdgcn_s_wait_asynccnt(0);
#endif
    }
    __syncthreads();                         // buffer i visible to all waves

    const bf16_t* as = As[i & 1];
    const bf16_t* bs = Bs[i & 1];
    bf16x16 bfrag[4];
#pragma unroll
    for (int tn = 0; tn < 4; ++tn) {
      // B layout used here (16-bit 32x16, contiguous k): lanes<16 hold K 0..15, lanes>=16 K 16..31
      const int nrow = wn * 64 + tn * 16 + (lane & 15);
      bfrag[tn] = *(const bf16x16*)(&bs[nrow * BK + half * 16]);
    }
#pragma unroll
    for (int tm = 0; tm < 2; ++tm) {
      // A layout (matches B): lanes<16 hold K 0..15,
      // lanes>=16 hold K 16..31
      const int mrow = wm * 32 + tm * 16 + (lane & 15);
      const bf16x8 alo = *(const bf16x8*)(&as[mrow * BK + half * 16]);
      const bf16x8 ahi = *(const bf16x8*)(&as[mrow * BK + half * 16 + 8]);
      const bf16x16 afrag = __builtin_shufflevector(alo, ahi,
          0, 1, 2, 3, 4, 5, 6, 7, 8, 9, 10, 11, 12, 13, 14, 15);
#pragma unroll
      for (int tn = 0; tn < 4; ++tn)
        acc[tm][tn] = __builtin_amdgcn_wmma_f32_16x16x32_bf16(
            false, afrag, false, bfrag[tn], (short)0, acc[tm][tn], false, false);
    }
    __syncthreads();                         // done reading buffer i
  }

  // epilogue: C/D layout -> VGPR r holds M = r + 8*(lane>>4), N = lane&15
#pragma unroll
  for (int tm = 0; tm < 2; ++tm)
#pragma unroll
    for (int tn = 0; tn < 4; ++tn)
#pragma unroll
      for (int r = 0; r < 8; ++r) {
        const int row = m0 + wm * 32 + tm * 16 + r + 8 * half;
        const int col = n0 + wn * 64 + tn * 16 + (lane & 15);
        const float a = acc[tm][tn][r];
        if (MODE == 0) {
          const float v = tanhf(a + aux[col]) * CODEC_RANGE;
          C[(size_t)row * N + col]  = v;
          C2[(size_t)row * N + col] = 0.0f;
          Cb[(size_t)row * N + col] = (bf16_t)v;
        } else if (MODE == 1) {
          C[(size_t)row * N + col] = aux[col] - 2.0f * a;   // ||c||^2 - 2 r.c
        } else {
          C[(size_t)row * N + col] = a + aux[col];
        }
      }
}

// ------------------------- f32 -> bf16 convert ------------------------------
__global__ __launch_bounds__(256)
void convert_bf16_kernel(const float* __restrict__ src, bf16_t* __restrict__ dst,
                         long long n) {
  const long long base = ((long long)blockIdx.x * 256 + threadIdx.x) * 8;
  if (base + 8 > n) return;
  const float4 a = *(const float4*)(src + base);
  const float4 b = *(const float4*)(src + base + 4);
  bf16x8 o;
  o[0] = (bf16_t)a.x; o[1] = (bf16_t)a.y; o[2] = (bf16_t)a.z; o[3] = (bf16_t)a.w;
  o[4] = (bf16_t)b.x; o[5] = (bf16_t)b.y; o[6] = (bf16_t)b.z; o[7] = (bf16_t)b.w;
  *(bf16x8*)(dst + base) = o;
}

// --------------- f32 [K][N] -> bf16 [N][K] transpose-convert ----------------
__global__ __launch_bounds__(256)
void transpose_bf16_kernel(const float* __restrict__ src, bf16_t* __restrict__ dst,
                           int K, int N) {
  __shared__ float tile[32][33];
  const int k0 = blockIdx.x * 32, n0 = blockIdx.y * 32, tid = threadIdx.x;
#pragma unroll
  for (int r = 0; r < 4; ++r) {
    const int kl = r * 8 + (tid >> 5), nl = tid & 31;
    tile[kl][nl] = src[(size_t)(k0 + kl) * N + n0 + nl];
  }
  __syncthreads();
#pragma unroll
  for (int r = 0; r < 4; ++r) {
    const int nl = r * 8 + (tid >> 5), kl = tid & 31;
    dst[(size_t)(n0 + nl) * K + k0 + kl] = (bf16_t)tile[kl][nl];
  }
}

// ------------- cb_sq[q][k] = ||cb[q][k]||^2 (wave per row), zero loss -------
__global__ __launch_bounds__(256)
void cbsq_kernel(const float* __restrict__ cb, float* __restrict__ cbsq,
                 float* __restrict__ loss) {
  if (blockIdx.x == 0 && threadIdx.x == 0) *loss = 0.0f;
  const int lane = threadIdx.x & 31;
  const int wrow = (blockIdx.x * 256 + threadIdx.x) >> 5;
  if (wrow >= NQ * KC) return;
  const float* row = cb + (size_t)wrow * DC;
  float s = 0.0f;
  for (int d = lane; d < DC; d += 32) { const float v = row[d]; s += v * v; }
#pragma unroll
  for (int off = 16; off; off >>= 1) s += __shfl_xor(s, off, 32);
  if (lane == 0) cbsq[wrow] = s;
}

// ---------------------------------------------------------------------------
// Per row: first-min argmin over K scores, gather codebook row, Q += c,
// R -= c (f32 master + bf16 mirror), commitment-loss partial via atomic.
// ---------------------------------------------------------------------------
__global__ __launch_bounds__(256)
void argmin_update_kernel(const float* __restrict__ S, const float* __restrict__ cb,
                          float* __restrict__ R, float* __restrict__ Q,
                          bf16_t* __restrict__ Rb,
                          float* __restrict__ codes_f, int* __restrict__ codes_i,
                          float* __restrict__ loss) {
  const int lane = threadIdx.x & 31;
  const int row  = blockIdx.x * 8 + (threadIdx.x >> 5);
  const float* s = S + (size_t)row * KC;

  float best = 3.402823466e38f;
  int   bidx = 0;
#pragma unroll
  for (int i = 0; i < KC / 32; ++i) {           // ascending index within lane
    const int j = lane + i * 32;
    const float v = s[j];
    if (v < best) { best = v; bidx = j; }
  }
#pragma unroll
  for (int off = 16; off; off >>= 1) {          // lexicographic (val, idx) min
    const float ov = __shfl_xor(best, off, 32);
    const int   oi = __shfl_xor(bidx, off, 32);
    if (ov < best || (ov == best && oi < bidx)) { best = ov; bidx = oi; }
  }
  if (lane == 0) {
    codes_f[row] = (float)bidx;
    codes_i[row] = bidx;
  }
  const float* c = cb + (size_t)bidx * DC;
  float lsum = 0.0f;
  for (int d = lane; d < DC; d += 32) {
    const float cv = c[d];
    const float rv = R[(size_t)row * DC + d];
    const float df = cv - rv;
    lsum += df * df;
    const float nr = rv - cv;
    Q[(size_t)row * DC + d] += cv;
    R[(size_t)row * DC + d]  = nr;
    Rb[(size_t)row * DC + d] = (bf16_t)nr;
  }
#pragma unroll
  for (int off = 16; off; off >>= 1) lsum += __shfl_xor(lsum, off, 32);
  if (lane == 0) atomicAdd(loss, lsum);
}

// ---------------------------------------------------------------------------
// sub[q][b][d][t] = cb[q][codes[q][b][t]][d]  via 32x32 LDS tile transpose.
// ---------------------------------------------------------------------------
__global__ __launch_bounds__(256)
void sub_gather_kernel(const float* __restrict__ cb, const int* __restrict__ codes_i,
                       float* __restrict__ sub) {
  __shared__ float tile[32][33];
  const int t0  = blockIdx.x * 32;
  const int dt  = blockIdx.y & 15;
  const int b   = blockIdx.y >> 4;
  const int q   = blockIdx.z;
  const int d0  = dt * 32;
  const int tid = threadIdx.x;
#pragma unroll
  for (int r = 0; r < 4; ++r) {
    const int tl = r * 8 + (tid >> 5);
    const int dl = tid & 31;
    const int t  = t0 + tl;
    float v = 0.0f;
    if (t < Tt) {
      const int idx = codes_i[q * MR + b * Tt + t];
      v = cb[((size_t)q * KC + idx) * DC + d0 + dl];
    }
    tile[tl][dl] = v;
  }
  __syncthreads();
#pragma unroll
  for (int r = 0; r < 4; ++r) {
    const int dl = r * 8 + (tid >> 5);
    const int tl = tid & 31;
    const int t  = t0 + tl;
    if (t < Tt)
      sub[(((size_t)q * Bb + b) * DC + d0 + dl) * (size_t)Tt + t] = tile[tl][dl];
  }
}

__global__ void loss_finalize_kernel(const float* __restrict__ loss,
                                     float* __restrict__ out) {
  *out = *loss / (float)((long long)NQ * MR * DC);
}

// ---------------------------------------------------------------------------
extern "C" void kernel_launch(void* const* d_in, const int* in_sizes, int n_in,
                              void* d_out, int out_size, void* d_ws, size_t ws_size,
                              hipStream_t stream) {
  (void)in_sizes; (void)n_in; (void)out_size; (void)ws_size;
  const float* x  = (const float*)d_in[0];   // [16,2000,1024]
  const float* Wi = (const float*)d_in[1];   // [1024,512]
  const float* bi = (const float*)d_in[2];   // [512]
  const float* Wo = (const float*)d_in[3];   // [512,1024]
  const float* bo = (const float*)d_in[4];   // [1024]
  const float* cb = (const float*)d_in[5];   // [8,1024,512]

  // d_out: out | codes | commit_loss | sub  (flat, return order)
  float* out       = (float*)d_out;
  float* codes_out = out + (size_t)MR * DIN;
  float* loss_out  = codes_out + (size_t)NQ * MR;
  float* sub_out   = loss_out + 1;

  // workspace layout (~372 MB): f32 masters, then bf16 mirrors
  char* w = (char*)d_ws;
  float* R    = (float*)w;  w += (size_t)MR * DC * 4;       // residual f32
  float* Q    = (float*)w;  w += (size_t)MR * DC * 4;       // quant_out f32
  float* S    = (float*)w;  w += (size_t)MR * KC * 4;       // VQ scores
  float* cbsq = (float*)w;  w += (size_t)NQ * KC * 4;
  int* codes_i = (int*)w;   w += (size_t)NQ * MR * 4;
  bf16_t* xb  = (bf16_t*)w; w += (size_t)MR * DIN * 2;      // x bf16; reused as Qb
  bf16_t* cbb = (bf16_t*)w; w += (size_t)NQ * KC * DC * 2;  // codebooks bf16
  bf16_t* Wib = (bf16_t*)w; w += (size_t)DC * DIN * 2;      // Wi^T bf16 [512][1024]
  bf16_t* Wob = (bf16_t*)w; w += (size_t)DIN * DC * 2;      // Wo^T bf16 [1024][512]
  bf16_t* Rb  = (bf16_t*)w; w += (size_t)MR * DC * 2;       // residual bf16 mirror
  float* loss = (float*)w;
  bf16_t* Qb  = xb;                                         // x dead after GEMM1

  // 0) operand preparation (bf16 packs, weight transposes, codebook norms)
  convert_bf16_kernel<<<(MR * DIN) / 2048, 256, 0, stream>>>(x, xb, (long long)MR * DIN);
  convert_bf16_kernel<<<(NQ * KC * DC) / 2048, 256, 0, stream>>>(cb, cbb, (long long)NQ * KC * DC);
  transpose_bf16_kernel<<<dim3(DIN / 32, DC / 32), 256, 0, stream>>>(Wi, Wib, DIN, DC);
  transpose_bf16_kernel<<<dim3(DC / 32, DIN / 32), 256, 0, stream>>>(Wo, Wob, DC, DIN);
  cbsq_kernel<<<(NQ * KC) / 8, 256, 0, stream>>>(cb, cbsq, loss);

  // 1) input projection: R = tanh(x@Wi + bi)*8 (f32 + bf16 mirror), Q = 0
  gemm_wmma_kernel<0><<<dim3(MR / BM, DC / BN), 256, 0, stream>>>(
      xb, Wib, bi, R, Q, Rb, MR, DC, DIN);

  // 2) residual quantization, sequential in q
  for (int q = 0; q < NQ; ++q) {
    gemm_wmma_kernel<1><<<dim3(MR / BM, KC / BN), 256, 0, stream>>>(
        Rb, cbb + (size_t)q * KC * DC, cbsq + q * KC, S, nullptr, nullptr,
        MR, KC, DC);
    argmin_update_kernel<<<MR / 8, 256, 0, stream>>>(
        S, cb + (size_t)q * KC * DC, R, Q, Rb,
        codes_out + (size_t)q * MR, codes_i + q * MR, loss);
  }

  // 3) output projection: out = Q@Wo + bo
  convert_bf16_kernel<<<(MR * DC) / 2048, 256, 0, stream>>>(Q, Qb, (long long)MR * DC);
  gemm_wmma_kernel<2><<<dim3(MR / BM, DIN / BN), 256, 0, stream>>>(
      Qb, Wob, bo, out, nullptr, nullptr, MR, DIN, DC);

  // 4) sub-quant gather/transpose and loss finalize
  sub_gather_kernel<<<dim3((Tt + 31) / 32, Bb * 16, NQ), 256, 0, stream>>>(
      cb, codes_i, sub_out);
  loss_finalize_kernel<<<1, 1, 0, stream>>>(loss, loss_out);
}
